// Node_sampler_46033459479298
// MI455X (gfx1250) — matrix operand order repaired and checked by reading, not compiled, not run on hardware
//
#include <hip/hip_runtime.h>
#include <hip/hip_bf16.h>

#define HID   192
#define H2    384
#define NPG   256
#define NGRAPH 1024
#define NTOT  (NGRAPH * NPG)      // 262144
#define EDG   4194304
#define KEEPK 231
#define LN_EPS 1e-5f

typedef __attribute__((ext_vector_type(16))) __bf16 v16bf;
typedef __attribute__((ext_vector_type(8)))  float  v8f;

union BFrag { v16bf v; uint4 q[2]; };
union CTile { v8f v; float f[8]; };

__device__ __forceinline__ unsigned short f2bf(float f) {
    unsigned int u = __float_as_uint(f);
    unsigned int r = (u + 0x7FFFu + ((u >> 16) & 1u)) >> 16;
    return (unsigned short)r;
}

// ---------------- Kernel 0: W1 [192][384] f32 -> W1T [384][192] bf16 ----------------
__global__ void conv_w1_kernel(const float* __restrict__ w1,
                               unsigned short* __restrict__ w1t) {
    int i = blockIdx.x * 256 + threadIdx.x;
    if (i < HID * H2) {
        int k = i / H2;
        int n = i - k * H2;
        w1t[n * HID + k] = f2bf(w1[i]);
    }
}

// ---------------- Kernel 1: fused (h+g) @ W1 -> LN -> ReLU -> dot(W2) -> logits -----
// Block: 256 threads (8 waves). Tile: 32 rows x 384 cols, K = 192 (6 steps of 32).
// Wave w: rows (w&1)*16..+15, cols (w>>2... ) -> (w>>1)*96..+95 (6 WMMA col tiles).
struct P0Buf {
    unsigned short a[32 * 200];   // x tile, bf16, row stride 200 (400B, 16B multiple)
    float gs[HID];
};
#define ZSTR 390                  // float stride; 8*390 % 64 == 48 -> no bank clash
union SMemU {
    P0Buf p0;
    float z[32 * ZSTR];
};

__global__ __launch_bounds__(256) void gemm_ln_logits_kernel(
    const float* __restrict__ h, const float* __restrict__ g,
    const unsigned short* __restrict__ w1t,
    const float* __restrict__ b1, const float* __restrict__ gamma,
    const float* __restrict__ beta, const float* __restrict__ w2,
    const float* __restrict__ b2p, float* __restrict__ logits)
{
    __shared__ __align__(16) SMemU sh;
    __shared__ float b1s[H2], gms[H2], bts[H2], w2s[H2];

    const int tid     = threadIdx.x;
    const int rowbase = blockIdx.x * 32;
    const int graph   = rowbase / NPG;   // 32 | 256 -> whole tile in one graph

    for (int i = tid; i < H2; i += 256) {
        b1s[i] = b1[i]; gms[i] = gamma[i]; bts[i] = beta[i]; w2s[i] = w2[i];
    }
    for (int i = tid; i < HID; i += 256) sh.p0.gs[i] = g[graph * HID + i];
    __syncthreads();

    // Stage x = h + g_rep as bf16 into LDS
    for (int i = tid; i < 32 * HID; i += 256) {
        int r = i / HID, k = i - r * HID;
        float v = h[(size_t)(rowbase + r) * HID + k] + sh.p0.gs[k];
        sh.p0.a[r * 200 + k] = f2bf(v);
    }
    __syncthreads();

    const int lane  = tid & 31;
    const int wave  = tid >> 5;
    const int rb    = (wave & 1) * 16;   // row half within tile
    const int cb    = (wave >> 1) * 96;  // column quarter (6 x 16 cols)
    const int lhalf = lane >> 4;         // 0: lanes 0-15, 1: lanes 16-31
    const int lmod  = lane & 15;

    CTile acc[6];
    #pragma unroll
    for (int t = 0; t < 6; ++t)
        #pragma unroll
        for (int r = 0; r < 8; ++r) acc[t].f[r] = 0.0f;

    #pragma unroll
    for (int ks = 0; ks < 6; ++ks) {
        const int K0 = ks * 32;
        // A fragment (16x32 bf16): lane<16 -> K{0..7,16..23}; lane>=16 -> K{8..15,24..31}
        BFrag af;
        const unsigned short* ap = &sh.p0.a[(rb + lmod) * 200 + K0 + lhalf * 8];
        af.q[0] = *(const uint4*)(ap);
        af.q[1] = *(const uint4*)(ap + 16);
        #pragma unroll
        for (int t = 0; t < 6; ++t) {
            // B fragment (32x16 bf16): lanes 0-15 -> K{0..7,16..23} of col lmod; 16-31 -> K{8..15,24..31}
            BFrag bf;
            const unsigned short* bp =
                &w1t[(size_t)(cb + t * 16 + lmod) * HID + K0 + lhalf * 8];
            bf.q[0] = *(const uint4*)(bp);
            bf.q[1] = *(const uint4*)(bp + 16);
            acc[t].v = __builtin_amdgcn_wmma_f32_16x16x32_bf16(
                false, af.v, false, bf.v, (short)0, acc[t].v, false, false);
        }
    }
    __syncthreads();   // all waves done reading A staging before z overwrites it

    // Spill C (f32 16x16 layout: VGPR r -> M=r+8*lhalf, lane -> N=lmod)
    #pragma unroll
    for (int t = 0; t < 6; ++t) {
        const int col = cb + t * 16 + lmod;
        #pragma unroll
        for (int r = 0; r < 8; ++r)
            sh.z[(rb + r + lhalf * 8) * ZSTR + col] = acc[t].f[r];
    }
    __syncthreads();

    // LayerNorm + ReLU + dot(W2): each wave handles 4 rows, lane covers 12 cols
    const float b2v = b2p[0];
    for (int lr = 0; lr < 4; ++lr) {
        const int row = wave * 4 + lr;
        float s = 0.f, sq = 0.f;
        #pragma unroll
        for (int j = 0; j < 12; ++j) {
            const int c = lane + j * 32;
            const float v = sh.z[row * ZSTR + c] + b1s[c];
            s += v; sq += v * v;
        }
        #pragma unroll
        for (int off = 16; off > 0; off >>= 1) {
            s  += __shfl_xor(s,  off, 32);
            sq += __shfl_xor(sq, off, 32);
        }
        const float mu   = s * (1.0f / H2);
        const float var  = sq * (1.0f / H2) - mu * mu;
        const float rstd = rsqrtf(var + LN_EPS);
        float dot = 0.f;
        #pragma unroll
        for (int j = 0; j < 12; ++j) {
            const int c = lane + j * 32;
            const float v  = sh.z[row * ZSTR + c] + b1s[c];
            float zn = (v - mu) * rstd * gms[c] + bts[c];
            zn = zn > 0.f ? zn : 0.f;
            dot += zn * w2s[c];
        }
        #pragma unroll
        for (int off = 16; off > 0; off >>= 1) dot += __shfl_xor(dot, off, 32);
        if (lane == 0) logits[rowbase + row] = dot + b2v;
    }
}

// ---------------- Kernel 2: per-graph top-K via in-LDS bitonic sort -----------------
__global__ __launch_bounds__(256) void topk_keep_kernel(
    const float* __restrict__ logits, float* __restrict__ keep,
    int* __restrict__ touched)
{
    __shared__ float vals[256];
    __shared__ int   idxs[256];
    const int tid   = threadIdx.x;
    const int gbase = blockIdx.x * NPG;

    vals[tid] = logits[gbase + tid];
    idxs[tid] = tid;
    touched[gbase + tid] = 0;   // zero scratch for edge-scatter pass
    __syncthreads();

    // Bitonic sort, descending by value, ascending index on ties (matches top_k)
    for (int k = 2; k <= 256; k <<= 1) {
        for (int j = k >> 1; j > 0; j >>= 1) {
            const int ixj = tid ^ j;
            if (ixj > tid) {
                float a = vals[tid], b = vals[ixj];
                int   ia = idxs[tid], ib = idxs[ixj];
                bool aAfter = (a < b) || (a == b && ia > ib); // a ranks later (desc)
                bool desc   = ((tid & k) == 0);
                if (desc ? aAfter : !aAfter) {
                    vals[tid] = b; vals[ixj] = a;
                    idxs[tid] = ib; idxs[ixj] = ia;
                }
            }
            __syncthreads();
        }
    }
    keep[gbase + idxs[tid]] = (tid < KEEPK) ? 1.0f : 0.0f;
}

// ---------------- Kernel 3: edge masking + weights + touched scatter ----------------
__global__ __launch_bounds__(256) void edge_kernel(
    const int* __restrict__ src, const int* __restrict__ dst,
    const float* __restrict__ keep, const float* __restrict__ logits,
    float* __restrict__ emask, float* __restrict__ ew, int* __restrict__ touched)
{
    int e = blockIdx.x * 256 + threadIdx.x;
    if (e >= EDG) return;
    int s = src[e], d = dst[e];
    float m = keep[s] * keep[d];
    emask[e] = m;
    ew[e]    = (logits[s] + logits[d]) * m;
    if (m != 0.0f) {
        atomicOr(&touched[s], 1);
        atomicOr(&touched[d], 1);
    }
}

// ---------------- Kernel 4: node_mask = touched > 0 ---------------------------------
__global__ __launch_bounds__(256) void node_mask_kernel(
    const int* __restrict__ touched, float* __restrict__ nm)
{
    int i = blockIdx.x * 256 + threadIdx.x;
    if (i < NTOT) nm[i] = touched[i] ? 1.0f : 0.0f;
}

// ------------------------------------------------------------------------------------
extern "C" void kernel_launch(void* const* d_in, const int* in_sizes, int n_in,
                              void* d_out, int out_size, void* d_ws, size_t ws_size,
                              hipStream_t stream)
{
    const float* h   = (const float*)d_in[0];
    const float* g   = (const float*)d_in[1];
    const int*   ei  = (const int*)  d_in[2];   // [2,E] flat: src then dst
    const float* W1  = (const float*)d_in[3];
    const float* b1  = (const float*)d_in[4];
    const float* gm  = (const float*)d_in[5];
    const float* bt  = (const float*)d_in[6];
    const float* W2  = (const float*)d_in[7];
    const float* b2  = (const float*)d_in[8];

    float* out   = (float*)d_out;
    float* emask = out;                               // E
    float* ew    = out + (size_t)EDG;                 // E
    float* lg    = out + 2 * (size_t)EDG;             // N (logits)
    float* nm    = lg + (size_t)NTOT;                 // N (node_mask)

    char* ws = (char*)d_ws;
    unsigned short* w1t = (unsigned short*)ws;                         // 147456 B
    float* keep    = (float*)(ws + 262144);                            // N floats
    int*   touched = (int*)  (ws + 262144 + (size_t)NTOT * 4);         // N ints

    conv_w1_kernel<<<(HID * H2 + 255) / 256, 256, 0, stream>>>(W1, w1t);
    gemm_ln_logits_kernel<<<NTOT / 32, 256, 0, stream>>>(
        h, g, w1t, b1, gm, bt, W2, b2, lg);
    topk_keep_kernel<<<NGRAPH, 256, 0, stream>>>(lg, keep, touched);
    edge_kernel<<<EDG / 256, 256, 0, stream>>>(
        ei, ei + EDG, keep, lg, emask, ew, touched);
    node_mask_kernel<<<NTOT / 256, 256, 0, stream>>>(touched, nm);
}
